// BatchProgramCC_6906307412257
// MI455X (gfx1250) — matrix operand order repaired and checked by reading, not compiled, not run on hardware
//
#include <hip/hip_runtime.h>
#include <cmath>

// ---------------- constants (from reference) ----------------
constexpr int V = 50000, E = 128, D = 256, H = 128, Lc = 2;
constexpr int Bc = 32, Sc = 128, Tc = 32;
constexpr int G3H = 3 * H;                 // 384
constexpr size_t ROWS = (size_t)Bc * Sc;   // 4096

// ---------------- WMMA types ----------------
typedef __bf16 bf16;
typedef __attribute__((ext_vector_type(16))) __bf16 v16bf;
typedef __attribute__((ext_vector_type(8)))  float  v8f;

union BFrag {
    v16bf v;
    uint4 q[2];
    bf16  e[16];
};

__device__ __forceinline__ v8f wmma_bf16(v16bf a, v16bf b, v8f c) {
    // D = A(16x32 bf16) x B(32x16 bf16) + C(16x16 f32)
    return __builtin_amdgcn_wmma_f32_16x16x32_bf16(
        /*neg_a=*/false, a, /*neg_b=*/false, b,
        /*c_mod=*/(short)0, c, /*reuse_a=*/false, /*reuse_b=*/false);
}

__device__ __forceinline__ float sigmoidf_(float x) {
    return 1.0f / (1.0f + __expf(-x));
}

// ---------------- prep: f32 -> bf16 ----------------
__global__ void k_cvt_bf16(const float* __restrict__ src, bf16* __restrict__ dst, int n) {
    int i = blockIdx.x * blockDim.x + threadIdx.x;
    if (i < n) dst[i] = (bf16)src[i];
}

// ---------------- K1: embed -> Wc -> tanh -> masked token max -> enc ----------------
// one block (128 thr = 4 waves) per (b, s). Output tile 32(tokens) x 256(D).
// Wc (64KB bf16) is staged into LDS once per block via GLOBAL_LOAD_ASYNC_TO_LDS_B128
// (ASYNCcnt path), so the WMMA K-loop feeds B fragments from LDS.
__global__ __launch_bounds__(128) void k_stmt(
    const int*  __restrict__ tokens,     // [B,S,T]
    const int*  __restrict__ tlens,      // [B,S]
    const int*  __restrict__ plens,      // [B]
    const bf16* __restrict__ embb,       // [V,E] bf16
    const bf16* __restrict__ wcb,        // [D,E] bf16 (row-major [N,K])
    const float* __restrict__ wc_bias,   // [D]
    float* __restrict__ enc_f,           // [B,S,D]
    bf16*  __restrict__ enc_b)           // [B,S,D]
{
    __shared__ __align__(16) bf16 lds_wc[D * E];   // 64 KB

    const int b = blockIdx.x / Sc, s = blockIdx.x % Sc;
    const int tid = threadIdx.x, lane = tid & 31, w = tid >> 5;
    const int khalf = lane >> 4, lr = lane & 15;

    float* encrow  = enc_f + (size_t)(b * Sc + s) * D;
    bf16*  encrowb = enc_b + (size_t)(b * Sc + s) * D;

    const bool valid = s >= (Sc - plens[b]);
    if (!valid) {   // zero-padded statement (uniform per block)
        for (int i = tid; i < D; i += 128) { encrow[i] = 0.0f; encrowb[i] = (bf16)0.0f; }
        return;
    }

    // --- async stage Wc -> LDS (65536 bytes, 16B per lane-op, 32 ops/thread) ---
    {
        const uint32_t lds_base = (uint32_t)(uintptr_t)(&lds_wc[0]);
        const char* gbase = (const char*)wcb;
        #pragma unroll 4
        for (int it = 0; it < 32; ++it) {
            const int byte_off = (tid + it * 128) * 16;
            uint64_t ga = (uint64_t)(uintptr_t)(gbase + byte_off);
            uint32_t la = lds_base + (uint32_t)byte_off;
            asm volatile("global_load_async_to_lds_b128 %0, %1, off"
                         :: "v"(la), "v"(ga) : "memory");
        }
        asm volatile("s_wait_asynccnt 0" ::: "memory");
    }

    const int tlen = tlens[b * Sc + s];
    const int* toks = tokens + (size_t)(b * Sc + s) * Tc;

    // A fragments: 2 M-tiles (token rows 0..15 / 16..31) x 4 K-tiles (E=128)
    BFrag afrag[2][4];
    for (int mt = 0; mt < 2; ++mt) {
        const int tok = toks[mt * 16 + lr];
        const bf16* rp = embb + (size_t)tok * E;
        for (int kt = 0; kt < 4; ++kt) {
            const bf16* pa = rp + kt * 32 + khalf * 8;
            afrag[mt][kt].q[0] = *(const uint4*)(pa);
            afrag[mt][kt].q[1] = *(const uint4*)(pa + 16);
        }
    }

    __syncthreads();   // LDS Wc visible to all waves

    // each wave: 4 N-tiles of D
    for (int nt = w * 4; nt < w * 4 + 4; ++nt) {
        v8f acc0 = {}, acc1 = {};
        const int dcol = nt * 16 + lr;
        const bf16* wrow = &lds_wc[dcol * E + khalf * 8];
        for (int kt = 0; kt < 4; ++kt) {
            BFrag bfr;
            const bf16* pb = wrow + kt * 32;
            bfr.q[0] = *(const uint4*)(pb);       // ds_load_b128
            bfr.q[1] = *(const uint4*)(pb + 16);
            acc0 = wmma_bf16(afrag[0][kt].v, bfr.v, acc0);
            acc1 = wmma_bf16(afrag[1][kt].v, bfr.v, acc1);
        }
        // epilogue: bias + tanh + masked max over the 32 token rows
        const float bias = wc_bias[dcol];
        float cmax = -2.0f;                      // tanh in (-1,1)
        for (int r = 0; r < 8; ++r) {
            const int row0 = r + khalf * 8;      // rows 0..15
            float v0 = tanhf(acc0[r] + bias);
            if (row0 < tlen) cmax = fmaxf(cmax, v0);
            const int row1 = 16 + row0;          // rows 16..31
            float v1 = tanhf(acc1[r] + bias);
            if (row1 < tlen) cmax = fmaxf(cmax, v1);
        }
        cmax = fmaxf(cmax, __shfl_xor(cmax, 16, 32));  // combine lane halves
        if (khalf == 0) { encrow[dcol] = cmax; encrowb[dcol] = (bf16)cmax; }
    }
}

// ---------------- K2: xw = enc @ Wih^T + bih ----------------
// one block per 16 rows of [B*S]; 4 waves x 6 N-tiles = 24 tiles (3H=384)
__global__ __launch_bounds__(128) void k_xw(
    const bf16*  __restrict__ encb,   // [B*S, D] bf16
    const bf16*  __restrict__ wih,    // [3H, D] bf16 (row-major [N,K])
    const float* __restrict__ bih,    // [3H]
    float* __restrict__ xw)           // [B*S, 3H]
{
    const int row0 = blockIdx.x * 16;
    const int tid = threadIdx.x, lane = tid & 31, w = tid >> 5;
    const int khalf = lane >> 4, lr = lane & 15;

    BFrag afrag[8];
    const bf16* arow = encb + (size_t)(row0 + lr) * D;
    for (int kt = 0; kt < 8; ++kt) {
        const bf16* pa = arow + kt * 32 + khalf * 8;
        afrag[kt].q[0] = *(const uint4*)(pa);
        afrag[kt].q[1] = *(const uint4*)(pa + 16);
    }
    for (int nt = w * 6; nt < w * 6 + 6; ++nt) {
        const int g = nt * 16 + lr;
        const bf16* wrow = wih + (size_t)g * D + khalf * 8;
        __builtin_prefetch(wrow, 0, 3);          // global_prefetch_b8
        v8f acc = {};
        for (int kt = 0; kt < 8; ++kt) {
            BFrag bfr;
            const bf16* pb = wrow + kt * 32;
            bfr.q[0] = *(const uint4*)(pb);
            bfr.q[1] = *(const uint4*)(pb + 16);
            acc = wmma_bf16(afrag[kt].v, bfr.v, acc);
        }
        const float bias = bih[g];
        for (int r = 0; r < 8; ++r) {
            const int row = row0 + r + khalf * 8;
            xw[(size_t)row * G3H + g] = acc[r] + bias;
        }
    }
}

// ---------------- K3: sequential GRU scan ----------------
// grid = 4 persistent blocks: (pair p, direction dir). Per step:
//   gh = h(32x128) @ Whh^T(128x384) + bhh  via WMMA from LDS, then fused gates.
__global__ __launch_bounds__(128) void k_gru(
    const float* __restrict__ xw_all,   // [4][B*S][3H]  (pd = p*2+dir)
    const bf16*  __restrict__ whh_all,  // [2][3H][H] bf16
    const float* __restrict__ bhh_fwd,  // [3H]
    const float* __restrict__ bhh_bwd,  // [3H]
    float* __restrict__ out_all)        // [4][B*S][H]
{
    const int pd = blockIdx.x;
    const int dir = pd & 1;
    const float* xw  = xw_all  + (size_t)pd * ROWS * G3H;
    const bf16*  whh = whh_all + (size_t)dir * G3H * H;
    const float* bhh = dir ? bhh_bwd : bhh_fwd;
    float* out = out_all + (size_t)pd * ROWS * H;

    const int tid = threadIdx.x, lane = tid & 31, w = tid >> 5;
    const int khalf = lane >> 4, lr = lane & 15;

    __shared__ float hbuf[32 * 128];    // h, f32 (16 KB)
    __shared__ float gh[32 * 384];      // Whh gates (48 KB)

    for (int i = tid; i < 32 * 128; i += 128) hbuf[i] = 0.0f;
    __syncthreads();

    for (int step = 0; step < Sc; ++step) {
        const int s_idx = dir ? (Sc - 1 - step) : step;

        // A fragments from LDS h (f32 -> bf16): 2 M-tiles x 4 K-tiles
        BFrag afrag[2][4];
        for (int mt = 0; mt < 2; ++mt) {
            const int row = mt * 16 + lr;
            const float* hr = &hbuf[row * 128 + khalf * 8];
            for (int kt = 0; kt < 4; ++kt) {
                #pragma unroll
                for (int i = 0; i < 8; ++i) {
                    afrag[mt][kt].e[i]     = (bf16)hr[kt * 32 + i];
                    afrag[mt][kt].e[8 + i] = (bf16)hr[kt * 32 + 16 + i];
                }
            }
        }
        // GEMM: each wave 6 N-tiles x 2 M-tiles, K=128 (4 wmma each)
        for (int nt = w * 6; nt < w * 6 + 6; ++nt) {
            const int g = nt * 16 + lr;
            const bf16* wrow = whh + (size_t)g * H + khalf * 8;
            v8f acc0 = {}, acc1 = {};
            for (int kt = 0; kt < 4; ++kt) {
                BFrag bfr;
                const bf16* pb = wrow + kt * 32;
                bfr.q[0] = *(const uint4*)(pb);
                bfr.q[1] = *(const uint4*)(pb + 16);
                acc0 = wmma_bf16(afrag[0][kt].v, bfr.v, acc0);
                acc1 = wmma_bf16(afrag[1][kt].v, bfr.v, acc1);
            }
            const float bias = bhh[g];
            for (int r = 0; r < 8; ++r) {
                const int rr = r + khalf * 8;
                gh[rr * G3H + g]        = acc0[r] + bias;
                gh[(16 + rr) * G3H + g] = acc1[r] + bias;
            }
        }
        __syncthreads();

        // fused gates: 32 batches x 128 hidden = 4096 elems / 128 threads
        for (int e2 = 0; e2 < 32; ++e2) {
            const int j = tid + 128 * e2;
            const int brow = j >> 7, hc = j & 127;
            const float* xr = xw + (size_t)(brow * Sc + s_idx) * G3H;
            const float ghr = gh[brow * G3H + hc];
            const float ghz = gh[brow * G3H + 128 + hc];
            const float ghn = gh[brow * G3H + 256 + hc];
            const float r_ = sigmoidf_(xr[hc] + ghr);
            const float z_ = sigmoidf_(xr[128 + hc] + ghz);
            const float n_ = tanhf(xr[256 + hc] + r_ * ghn);
            const float h_ = hbuf[j];
            const float hn = (1.0f - z_) * n_ + z_ * h_;
            hbuf[j] = hn;
            out[(size_t)(brow * Sc + s_idx) * H + hc] = hn;   // backward pre-reversed
        }
        __syncthreads();
    }
}

// ---------------- K4: residual + time max-pool + |d| + linear + softmax ----------------
__global__ __launch_bounds__(256) void k_head(
    const float* __restrict__ enc_f,    // [2][B*S][D]
    const float* __restrict__ out_all,  // [4][B*S][H]
    const float* __restrict__ h2l_w,    // [L, 2H]
    const float* __restrict__ h2l_b,    // [L]
    float* __restrict__ outp)           // [B, L, 1]
{
    const int b = blockIdx.x, d = threadIdx.x;
    float v[2];
    for (int p = 0; p < 2; ++p) {
        const float* encp = enc_f   + (size_t)p * ROWS * D;
        const float* of   = out_all + (size_t)(p * 2 + 0) * ROWS * H;
        const float* ob   = out_all + (size_t)(p * 2 + 1) * ROWS * H;
        float m = -1e30f;
        for (int s = 0; s < Sc; ++s) {
            const size_t row = (size_t)b * Sc + s;
            float g = (d < H) ? of[row * H + d] : ob[row * H + (d - H)];
            g += encp[row * D + d];
            m = fmaxf(m, g);
        }
        v[p] = m;
    }
    const float ad = fabsf(v[0] - v[1]);
    __shared__ float red[2][256];
    red[0][d] = ad * h2l_w[d];
    red[1][d] = ad * h2l_w[256 + d];
    __syncthreads();
    for (int off = 128; off > 0; off >>= 1) {
        if (d < off) { red[0][d] += red[0][d + off]; red[1][d] += red[1][d + off]; }
        __syncthreads();
    }
    if (d == 0) {
        const float l0 = red[0][0] + h2l_b[0];
        const float l1 = red[1][0] + h2l_b[1];
        const float mm = fmaxf(l0, l1);
        const float e0 = __expf(l0 - mm), e1 = __expf(l1 - mm);
        const float inv = 1.0f / (e0 + e1);
        outp[b * 2 + 0] = e0 * inv;
        outp[b * 2 + 1] = e1 * inv;
    }
}

// ---------------- host ----------------
extern "C" void kernel_launch(void* const* d_in, const int* in_sizes, int n_in,
                              void* d_out, int out_size, void* d_ws, size_t ws_size,
                              hipStream_t stream) {
    (void)in_sizes; (void)n_in; (void)out_size; (void)ws_size;

    const int*   tokens1 = (const int*)d_in[0];
    const int*   tlens1  = (const int*)d_in[1];
    const int*   plens1  = (const int*)d_in[2];
    const int*   tokens2 = (const int*)d_in[3];
    const int*   tlens2  = (const int*)d_in[4];
    const int*   plens2  = (const int*)d_in[5];
    const float* emb     = (const float*)d_in[6];
    const float* Wc_w    = (const float*)d_in[7];
    const float* Wc_b    = (const float*)d_in[8];
    const float* wih_f   = (const float*)d_in[9];
    const float* whh_f   = (const float*)d_in[10];
    const float* bih_f   = (const float*)d_in[11];
    const float* bhh_f   = (const float*)d_in[12];
    const float* wih_b   = (const float*)d_in[13];
    const float* whh_b   = (const float*)d_in[14];
    const float* bih_b   = (const float*)d_in[15];
    const float* bhh_b   = (const float*)d_in[16];
    const float* h2l_w   = (const float*)d_in[17];
    const float* h2l_b   = (const float*)d_in[18];
    float* outp = (float*)d_out;

    // workspace carve (256B aligned)
    uint8_t* base = (uint8_t*)d_ws;
    size_t off = 0;
    auto take = [&](size_t bytes) -> void* {
        void* r = base + off;
        off += (bytes + 255) & ~(size_t)255;
        return r;
    };
    const size_t N_EMB = (size_t)V * E;       // 6,400,000
    const size_t N_WC  = (size_t)D * E;       // 32,768
    const size_t N_WIH = (size_t)G3H * D;     // 98,304
    const size_t N_WHH = (size_t)G3H * H;     // 49,152

    bf16*  emb_bf  = (bf16*)take(N_EMB * sizeof(bf16));
    bf16*  wc_bf   = (bf16*)take(N_WC * sizeof(bf16));
    bf16*  wih_bf  = (bf16*)take(2 * N_WIH * sizeof(bf16));
    bf16*  whh_bf  = (bf16*)take(2 * N_WHH * sizeof(bf16));
    float* enc_f   = (float*)take(2 * ROWS * D * sizeof(float));
    bf16*  enc_b   = (bf16*)take(2 * ROWS * D * sizeof(bf16));
    float* xw      = (float*)take(4 * ROWS * G3H * sizeof(float));
    float* out_all = (float*)take(4 * ROWS * H * sizeof(float));

    // --- prep: bf16 conversions ---
    auto cvt = [&](const float* s, bf16* dst, size_t n) {
        k_cvt_bf16<<<(unsigned)((n + 255) / 256), 256, 0, stream>>>(s, dst, (int)n);
    };
    cvt(emb,   emb_bf,          N_EMB);
    cvt(Wc_w,  wc_bf,           N_WC);
    cvt(wih_f, wih_bf,          N_WIH);
    cvt(wih_b, wih_bf + N_WIH,  N_WIH);
    cvt(whh_f, whh_bf,          N_WHH);
    cvt(whh_b, whh_bf + N_WHH,  N_WHH);

    // --- K1: statement encodings for both programs ---
    k_stmt<<<(unsigned)ROWS, 128, 0, stream>>>(tokens1, tlens1, plens1, emb_bf, wc_bf, Wc_b,
                                               enc_f, enc_b);
    k_stmt<<<(unsigned)ROWS, 128, 0, stream>>>(tokens2, tlens2, plens2, emb_bf, wc_bf, Wc_b,
                                               enc_f + ROWS * D, enc_b + ROWS * D);

    // --- K2: input-gate GEMMs, 4 slabs (p, dir) ---
    for (int p = 0; p < 2; ++p) {
        for (int dir = 0; dir < 2; ++dir) {
            const int pd = p * 2 + dir;
            k_xw<<<(unsigned)(ROWS / 16), 128, 0, stream>>>(
                enc_b + (size_t)p * ROWS * D,
                wih_bf + (size_t)dir * N_WIH,
                dir ? bih_b : bih_f,
                xw + (size_t)pd * ROWS * G3H);
        }
    }

    // --- K3: 4 persistent GRU scans ---
    k_gru<<<4, 128, 0, stream>>>(xw, whh_bf, bhh_f, bhh_b, out_all);

    // --- K4: head ---
    k_head<<<Bc, 256, 0, stream>>>(enc_f, out_all, h2l_w, h2l_b, outp);
}
